// KGAtt_13348758356215
// MI455X (gfx1250) — matrix operand order repaired and checked by reading, not compiled, not run on hardware
//
#include <hip/hip_runtime.h>
#include <math.h>

typedef __attribute__((ext_vector_type(16))) _Float16 v16h;
typedef __attribute__((ext_vector_type(8)))  float    v8f;
typedef __attribute__((ext_vector_type(4)))  float    v4f;   // trivial vector: AS1-loadable
typedef __attribute__((ext_vector_type(2)))  float    v2f;

// Address-space(1) (global) pointer for gathers: guarantees global_load_b128
// instead of flat_load_b128 (flat ties up DScnt + LDS arbitration on CDNA5).
typedef __attribute__((address_space(1))) const v4f* gf4ptr;

#define N_ENT   100000
#define N_REL   500
#define N_EDGES 800000
#define IN_DIM  128
#define K_DIM   384   // 3 * IN_DIM
#define OUT_DIM 128

// ---------------------------------------------------------------------------
// Kernel 0: zero the accumulation buffers (hs lives in d_out, ebs in d_ws)
// ---------------------------------------------------------------------------
__global__ void kg_zero(float* __restrict__ hs, float* __restrict__ ebs) {
    const long stride = (long)gridDim.x * blockDim.x;
    long idx = (long)blockIdx.x * blockDim.x + threadIdx.x;
    const long total = (long)N_ENT * OUT_DIM;
    for (long i = idx; i < total; i += stride) hs[i] = 0.0f;
    for (long i = idx; i < N_ENT; i += stride) ebs[i] = 0.0f;
}

// ---------------------------------------------------------------------------
// Kernel 1: per-edge GEMM (WMMA f16) + attention scalar + segment-sum atomics
//   block = 256 threads = 8 waves; each wave handles 16 edges.
//   Wa (128x384) is converted to f16 and staged in LDS once per block.
// ---------------------------------------------------------------------------
__global__ __launch_bounds__(256, 1) void kg_edge(
    const int*   __restrict__ trip,   // (N_EDGES, 3) head, tail, rel
    const float* __restrict__ ent,    // (N_ENT, 128)
    const float* __restrict__ rele,   // (N_REL, 128)
    const float* __restrict__ Wa,     // (128, 384)
    const float* __restrict__ ba,     // (128,)
    const float* __restrict__ Wa2,    // (1, 128)
    const float* __restrict__ ba2,    // (1,)
    float* __restrict__ hs,           // (N_ENT, 128) accum
    float* __restrict__ ebs)          // (N_ENT,) accum
{
    extern __shared__ _Float16 sWa[] __attribute__((aligned(32))); // [128][384] f16 = 96 KB

    // ---- stage Wa -> LDS as f16 (contiguous f32 pairs -> f16 pairs) ----
    for (int i = threadIdx.x; i < OUT_DIM * K_DIM / 2; i += 256) {
        v2f w = ((const v2f*)Wa)[i];
        sWa[2 * i + 0] = (_Float16)w[0];
        sWa[2 * i + 1] = (_Float16)w[1];
    }
    __syncthreads();

    const int lane   = threadIdx.x & 31;
    const int wave   = threadIdx.x >> 5;
    const int col    = lane & 15;   // A row (M) for loads / C column (N)
    const int halfId = lane >> 4;

    const long ebase = (long)blockIdx.x * 128 + (long)wave * 16;

    // Per-lane A-row sources: edge = ebase + col, feature = [ent[h] | rel[r] | ent[t]]
    const long eA = ebase + col;
    const int hA = trip[eA * 3 + 0];
    const int tA = trip[eA * 3 + 1];
    const int rA = trip[eA * 3 + 2];
    // Keep raw addresses as integers; each load goes through an explicit
    // global-addrspace pointer so it lowers to global_load_b128.
    const unsigned long long segAddr[3] = {
        (unsigned long long)(ent  + (long)hA * IN_DIM),
        (unsigned long long)(rele + (long)rA * IN_DIM),
        (unsigned long long)(ent  + (long)tA * IN_DIM)
    };

    v8f acc[8];
    {
        v8f z = {};
        #pragma unroll
        for (int t = 0; t < 8; ++t) acc[t] = z;
    }

    // ---- K loop: 12 tiles of 32 over the 384-long concat feature ----
    for (int kt = 0; kt < K_DIM / 32; ++kt) {
        const int k0 = kt * 32;

        // A fragment (16x32 f16): lane holds K = k0+half*16 .. +15
        const int kA0 = k0 + halfId * 16;
        const int kA1 = kA0 + 8;
        // 8-element runs never cross a 128-boundary segment of the concat
        gf4ptr p0 = (gf4ptr)(segAddr[kA0 >> 7] + (unsigned long long)(kA0 & 127) * sizeof(float));
        gf4ptr p1 = (gf4ptr)(segAddr[kA1 >> 7] + (unsigned long long)(kA1 & 127) * sizeof(float));
        v4f x0 = p0[0];
        v4f x1 = p0[1];
        v4f y0 = p1[0];
        v4f y1 = p1[1];
        v16h a;
        a[0]  = (_Float16)x0[0]; a[1]  = (_Float16)x0[1]; a[2]  = (_Float16)x0[2]; a[3]  = (_Float16)x0[3];
        a[4]  = (_Float16)x1[0]; a[5]  = (_Float16)x1[1]; a[6]  = (_Float16)x1[2]; a[7]  = (_Float16)x1[3];
        a[8]  = (_Float16)y0[0]; a[9]  = (_Float16)y0[1]; a[10] = (_Float16)y0[2]; a[11] = (_Float16)y0[3];
        a[12] = (_Float16)y1[0]; a[13] = (_Float16)y1[1]; a[14] = (_Float16)y1[2]; a[15] = (_Float16)y1[3];

        // B fragments from LDS: lane holds column n, K = k0 + half*16 .. +15
        const int kB0 = k0 + halfId * 16;
        #pragma unroll
        for (int nt = 0; nt < 8; ++nt) {
            const int n = nt * 16 + col;
            v16h b = *(const v16h*)(sWa + n * K_DIM + kB0);
            acc[nt] = __builtin_amdgcn_wmma_f32_16x16x32_f16(
                false, a, false, b, (short)0, acc[nt], false, false);
        }
    }

    // ---- add bias ba[n]: same for every row r in a C tile ----
    #pragma unroll
    for (int nt = 0; nt < 8; ++nt) {
        const float bn = ba[nt * 16 + col];
        #pragma unroll
        for (int r = 0; r < 8; ++r) acc[nt][r] += bn;
    }

    // ---- attention scalar: b = leaky_relu(c . Wa2 + ba2); e_b = exp(b) ----
    float w2v[8];
    #pragma unroll
    for (int nt = 0; nt < 8; ++nt) w2v[nt] = Wa2[nt * 16 + col];
    const float bias2 = ba2[0];

    float eb[8];  // e_b for the 8 edges (m = halfId*8 + r) this half owns
    #pragma unroll
    for (int r = 0; r < 8; ++r) {
        float p = 0.0f;
        #pragma unroll
        for (int nt = 0; nt < 8; ++nt) p += acc[nt][r] * w2v[nt];
        // reduce over the 16 lanes of this half (bit 4 untouched keeps halves apart)
        p += __shfl_xor(p, 1, 32);
        p += __shfl_xor(p, 2, 32);
        p += __shfl_xor(p, 4, 32);
        p += __shfl_xor(p, 8, 32);
        float bb = p + bias2;
        bb = bb > 0.0f ? bb : 0.01f * bb;   // jax.nn.leaky_relu default slope
        eb[r] = expf(bb);
    }

    // ---- segment-sum atomics keyed by head ----
    int hh[8];
    #pragma unroll
    for (int r = 0; r < 8; ++r)
        hh[r] = trip[(ebase + halfId * 8 + r) * 3];   // head of edge m = half*8 + r

    if (col == 0) {  // one lane per half handles the scalar denominator
        #pragma unroll
        for (int r = 0; r < 8; ++r) atomicAdd(&ebs[hh[r]], eb[r]);
    }
    #pragma unroll
    for (int r = 0; r < 8; ++r) {
        const float s = eb[r];
        float* dst = hs + (long)hh[r] * OUT_DIM + col;
        #pragma unroll
        for (int nt = 0; nt < 8; ++nt)
            atomicAdd(dst + nt * 16, s * acc[nt][r]);
    }
}

// ---------------------------------------------------------------------------
// Kernel 2: out = elu(hs / where(ebs==0, 1e-12, ebs)), in place in d_out
// ---------------------------------------------------------------------------
__global__ void kg_final(float* __restrict__ hs, const float* __restrict__ ebs) {
    const long i = (long)blockIdx.x * blockDim.x + threadIdx.x;
    if (i >= (long)N_ENT * OUT_DIM) return;
    float d = ebs[i >> 7];
    d = (d == 0.0f) ? 1e-12f : d;
    const float v = hs[i] / d;
    hs[i] = (v > 0.0f) ? v : (expf(v) - 1.0f);
}

extern "C" void kernel_launch(void* const* d_in, const int* in_sizes, int n_in,
                              void* d_out, int out_size, void* d_ws, size_t ws_size,
                              hipStream_t stream) {
    const int*   trip = (const int*)  d_in[0];
    const float* ent  = (const float*)d_in[1];
    const float* rele = (const float*)d_in[2];
    const float* Wa   = (const float*)d_in[3];
    const float* ba   = (const float*)d_in[4];
    const float* Wa2  = (const float*)d_in[5];
    const float* ba2  = (const float*)d_in[6];

    float* hs  = (float*)d_out;          // (N_ENT, 128) accumulators, finalized in place
    float* ebs = (float*)d_ws;           // (N_ENT,) denominator accumulators

    kg_zero<<<2048, 256, 0, stream>>>(hs, ebs);

    const size_t shmem = (size_t)OUT_DIM * K_DIM * sizeof(_Float16);  // 96 KB f16 Wa
    kg_edge<<<N_EDGES / 128, 256, shmem, stream>>>(trip, ent, rele, Wa, ba, Wa2, ba2, hs, ebs);

    kg_final<<<((long)N_ENT * OUT_DIM + 255) / 256, 256, 0, stream>>>(hs, ebs);
}
